// Sea_Attention_22144851378352
// MI455X (gfx1250) — compile-verified, matrix operand order checked by reading, not run on hardware
//
#include <hip/hip_runtime.h>
#include <hip/hip_bf16.h>

typedef __attribute__((ext_vector_type(16))) _Float16 v16h;
typedef __attribute__((ext_vector_type(8)))  float    v8f;

static constexpr int Bn   = 16;
static constexpr int DIM  = 256;
static constexpr int Hh   = 64;
static constexpr int Ww   = 64;
static constexpr int HW   = Hh * Ww;       // 4096
static constexpr int NPIX = Bn * HW;       // 65536
static constexpr int NH   = 8;
static constexpr int DH   = 256;           // 2 * NH * KD
static constexpr int QC   = 2 * DH;        // 512

__device__ __forceinline__ float relu6f(float v) {
    return v < 0.f ? 0.f : (v > 6.f ? 6.f : v);
}

// ---------------------------------------------------------------------------
// CDNA5 async global->LDS copy (16B per enabled lane, tracked by ASYNCcnt)
// ---------------------------------------------------------------------------
__device__ __forceinline__ void async_copy_b128(unsigned lds_off, const void* gaddr) {
    asm volatile("global_load_async_to_lds_b128 %0, %1, off"
                 :: "v"(lds_off), "v"(gaddr) : "memory");
}
__device__ __forceinline__ void wait_async0() {
    asm volatile("s_wait_asynccnt 0" ::: "memory");
}

// B-matrix packed fragment layout: element (n,k) of the [K,N] matrix lives at
//   [((k/32)*(N/16) + n/16)*16 + (n%16)]*32 + (k%32)
// so a lane's WMMA B fragment is two contiguous 16B runs (like the A layout).
__device__ __forceinline__ size_t bpidx(int n, int k, int N) {
    return ((size_t)((k >> 5) * (N >> 4) + (n >> 4)) * 16 + (n & 15)) * 32 + (k & 31);
}

union Frag { uint4 q[2]; v16h v; };

// ---------------------------------------------------------------------------
// WMMA GEMM:  D[M,N] = A[M,K] (f16, K-contig) x Bp (f16, fragment-packed)
// Block: 256 threads (8 waves). Block tile M=16, N=128 (one 16x16 tile/wave).
// A staged through LDS with global_load_async_to_lds_b128; B fragments read
// directly from global (weights are L2-resident).
// EPI 0: out[m*N+n] = acc + bias[n]
// EPI 1: out is [B,DIM,H,W]; val = h_sigmoid(acc+bias) * mul[m*DIM+n]
// ---------------------------------------------------------------------------
template <int EPI>
__global__ void gemm_wmma_f16(const _Float16* __restrict__ A,
                              const _Float16* __restrict__ Bp,
                              const float*    __restrict__ bias,
                              float*          __restrict__ out,
                              const float*    __restrict__ mul,
                              int M, int N, int K)
{
    __shared__ _Float16 sA[16][32];   // one 16x32 A tile per K-step

    const int tid   = threadIdx.x;
    const int wave  = tid >> 5;
    const int lane  = tid & 31;
    const int m0    = blockIdx.y * 16;
    const int n0    = blockIdx.x * 128;
    const int khalf = lane >> 4;                 // K half-group of this lane
    const int mrow  = lane & 15;                 // A fragment row
    const int ncol  = lane & 15;                 // B fragment column within tile
    const int nt    = (n0 >> 4) + wave;          // global n-tile index

    v8f acc = {};

    for (int k0 = 0; k0 < K; k0 += 32) {
        // ---- stage A tile: 64 async b128 transfers (lanes of waves 0..1) ----
        if (tid < 64) {
            int r  = tid >> 2;            // row 0..15
            int c8 = (tid & 3) * 8;       // col 0,8,16,24
            unsigned lds = (unsigned)(unsigned long long)&sA[r][c8];
            const void* g = (const void*)(A + (size_t)(m0 + r) * K + (k0 + c8));
            async_copy_b128(lds, g);
        }
        if (k0 + 32 < K) {  // pull next B slab toward L2 (global_prefetch)
            __builtin_prefetch(Bp + (((size_t)((k0 >> 5) + 1) * (N >> 4) + nt) * 16) * 32, 0, 0);
        }
        wait_async0();
        __syncthreads();

        // ---- fragments: A from LDS (2x ds_load_b128), B from global ----
        Frag fa, fb;
        fa.q[0] = *(const uint4*)(&sA[mrow][8 * khalf]);
        fa.q[1] = *(const uint4*)(&sA[mrow][16 + 8 * khalf]);
        const uint4* bq4 = (const uint4*)(Bp +
            (((size_t)(k0 >> 5) * (N >> 4) + nt) * 16 + ncol) * 32);
        fb.q[0] = bq4[khalf];
        fb.q[1] = bq4[2 + khalf];

        acc = __builtin_amdgcn_wmma_f32_16x16x32_f16(
                  /*neg_a=*/false, fa.v, /*neg_b=*/false, fb.v,
                  /*c_mod=*/(short)0, acc, /*reuse_a=*/false, /*reuse_b=*/false);
        __syncthreads();
    }

    // C/D layout: VGPR r -> M = r + 8*(lane/16), N = lane%16 (+ tile offsets)
    const int gn = n0 + wave * 16 + ncol;
#pragma unroll
    for (int r = 0; r < 8; ++r) {
        int   gm = m0 + r + 8 * khalf;
        float v  = acc[r] + bias[gn];
        if (EPI == 0) {
            out[(size_t)gm * N + gn] = v;
        } else {
            float s   = relu6f(v + 3.0f) * (1.0f / 6.0f);       // h_sigmoid
            float res = s * mul[(size_t)gm * DIM + gn];
            int   b_  = gm >> 12;        // HW = 4096
            int   hw  = gm & 4095;
            out[(size_t)b_ * DIM * HW + (size_t)gn * HW + hw] = res;
        }
    }
}

// ---------------------------------------------------------------------------
// x [B,DIM,H,W] f32  ->  xbf [n=B*HW, DIM] f16  (GEMM A layout, K contiguous)
// ---------------------------------------------------------------------------
__global__ void k_convert_x(const float* __restrict__ x, _Float16* __restrict__ xbf)
{
    size_t t  = (size_t)blockIdx.x * 256 + threadIdx.x;   // < NPIX*DIM
    int    hw = (int)(t % HW);
    size_t r  = t / HW;
    int    c  = (int)(r % DIM);
    int    b  = (int)(r / DIM);
    xbf[((size_t)(b * HW + hw)) * DIM + c] = (_Float16)x[t];
}

// Wq[128,256], Wk[128,256], Wv[256,256] -> fragment-packed wqkv (N=512, K=256)
__global__ void k_pack_wqkv(const float* __restrict__ Wq, const float* __restrict__ Wk,
                            const float* __restrict__ Wv, _Float16* __restrict__ dst)
{
    int t  = blockIdx.x * 256 + threadIdx.x;  // < 512*256
    int ci = t % DIM;
    int co = t / DIM;
    float v = (co < 128) ? Wq[co * DIM + ci]
            : (co < 256) ? Wk[(co - 128) * DIM + ci]
                         : Wv[(co - 256) * DIM + ci];
    dst[bpidx(co, ci, QC)] = (_Float16)v;
}

__global__ void k_pack_bias(const float* bq, const float* bk, const float* bv, float* dst)
{
    int t = blockIdx.x * 256 + threadIdx.x;
    if (t >= QC) return;
    dst[t] = (t < 128) ? bq[t] : (t < 256) ? bk[t - 128] : bv[t - 256];
}

// src [Nrows, K] f32 -> fragment-packed dst (B matrix [K][Nrows])
__global__ void k_pack_w(const float* __restrict__ src, _Float16* __restrict__ dst,
                         int Nrows, int K)
{
    int t = blockIdx.x * 256 + threadIdx.x;
    if (t >= Nrows * K) return;
    int k = t % K;
    int n = t / K;
    dst[bpidx(n, k, Nrows)] = (_Float16)src[t];
}

// depthwise 3x3 + relu6 on qkv [n, 512] f32 -> hbf [n, 512] f16
__global__ void k_dwconv(const float* __restrict__ qkv, const float* __restrict__ Wdw,
                         const float* __restrict__ bdw, _Float16* __restrict__ hbf)
{
    size_t t  = (size_t)blockIdx.x * 256 + threadIdx.x;   // < NPIX*QC
    int    c  = (int)(t % QC);
    size_t n  = t / QC;
    int    hw = (int)(n % HW);
    int    b  = (int)(n / HW);
    int    h  = hw >> 6, w = hw & 63;
    float  s  = bdw[c];
#pragma unroll
    for (int kh = 0; kh < 3; ++kh) {
        int hh = h + kh - 1;
        if (hh < 0 || hh >= Hh) continue;
#pragma unroll
        for (int kw = 0; kw < 3; ++kw) {
            int ww = w + kw - 1;
            if (ww < 0 || ww >= Ww) continue;
            s += Wdw[c * 9 + kh * 3 + kw] *
                 qkv[((size_t)b * HW + hh * 64 + ww) * QC + c];
        }
    }
    hbf[t] = (_Float16)relu6f(s);
}

// means over W (mrow) and over H (mcol) of qkv: [B,512,64] each
__global__ void k_mean(const float* __restrict__ qkv, float* __restrict__ mrow,
                       float* __restrict__ mcol)
{
    int t = blockIdx.x * 256 + threadIdx.x;   // < B*QC*64
    int p = t & 63;
    int c = (t >> 6) % QC;
    int b = t / (QC * 64);
    float sr = 0.f, sc = 0.f;
    for (int u = 0; u < 64; ++u) {
        sr += qkv[((size_t)b * HW + p * 64 + u) * QC + c];   // h=p, sum over w
        sc += qkv[((size_t)b * HW + u * 64 + p) * QC + c];   // w=p, sum over h
    }
    mrow[t] = sr * (1.0f / 64.0f);
    mcol[t] = sc * (1.0f / 64.0f);
}

// linear interp of pe [128,16] -> [128,64] (align_corners=False)
__global__ void k_peinterp(const float* __restrict__ pe, float* __restrict__ dst)
{
    int t = blockIdx.x * 256 + threadIdx.x;   // < 128*64
    int i = t & 63;
    int c = t >> 6;
    float src = (i + 0.5f) * 0.25f - 0.5f;
    src = src < 0.f ? 0.f : (src > 15.f ? 15.f : src);
    int   lo = (int)src;
    int   hi = lo + 1 > 15 ? 15 : lo + 1;
    float w  = src - (float)lo;
    dst[t] = pe[c * 16 + lo] * (1.0f - w) + pe[c * 16 + hi] * w;
}

// one block per (b, head): 64x64 attention with kd=16, d=32
__global__ void k_axial(const float* __restrict__ m, const float* __restrict__ peq,
                        const float* __restrict__ pek, float* __restrict__ xx)
{
    __shared__ float sq[16][64], sk[16][64], sv[32][64];
    int bh   = blockIdx.x;
    int head = bh % NH;
    int b    = bh / NH;
    int tid  = threadIdx.x;   // 64

    for (int e = tid; e < 16 * 64; e += 64) {
        int kd = e >> 6, i = e & 63;
        int c  = head * 16 + kd;
        sq[kd][i] = m[((size_t)b * QC + c) * 64 + i]       + peq[c * 64 + i];
        sk[kd][i] = m[((size_t)b * QC + 128 + c) * 64 + i] + pek[c * 64 + i];
    }
    for (int e = tid; e < 32 * 64; e += 64) {
        int d = e >> 6, j = e & 63;
        sv[d][j] = m[((size_t)b * QC + 256 + head * 32 + d) * 64 + j];
    }
    __syncthreads();

    int   i  = tid;
    float lg[64];
    float mx = -1e30f;
    for (int j = 0; j < 64; ++j) {
        float s = 0.f;
#pragma unroll
        for (int kd = 0; kd < 16; ++kd) s += sq[kd][i] * sk[kd][j];
        s *= 0.25f;                       // KD^-0.5
        lg[j] = s;
        mx = s > mx ? s : mx;
    }
    float sum = 0.f;
    for (int j = 0; j < 64; ++j) {
        float e2 = __expf(lg[j] - mx);
        lg[j] = e2;
        sum += e2;
    }
    float inv = 1.0f / sum;
    for (int d = 0; d < 32; ++d) {
        float acc = 0.f;
        for (int j = 0; j < 64; ++j) acc += lg[j] * sv[d][j];
        xx[((size_t)b * DH + head * 32 + d) * 64 + i] = acc * inv;
    }
}

// out[b,co,i] = bias[co] + sum_ci W[co,ci] * relu6(xx[b,ci,i])   ([B,256,64])
__global__ void k_smallconv(const float* __restrict__ xx, const float* __restrict__ Wt,
                            const float* __restrict__ bias, float* __restrict__ out)
{
    int t  = blockIdx.x * 256 + threadIdx.x;  // < B*DH*64
    int i  = t & 63;
    int co = (t >> 6) % DH;
    int b  = t / (DH * 64);
    float s = bias[co];
    for (int ci = 0; ci < DH; ++ci)
        s += Wt[co * DH + ci] * relu6f(xx[((size_t)b * DH + ci) * 64 + i]);
    out[t] = s;
}

// abf[n,c] = f16(relu6(v[n,c] + rowo[b,c,h] + colo[b,c,w]))
__global__ void k_aprep(const float* __restrict__ qkv, const float* __restrict__ rowo,
                        const float* __restrict__ colo, _Float16* __restrict__ abf)
{
    size_t t  = (size_t)blockIdx.x * 256 + threadIdx.x;   // < NPIX*DH
    int    c  = (int)(t % DH);
    size_t n  = t / DH;
    int    hw = (int)(n % HW);
    int    b  = (int)(n / HW);
    int    h  = hw >> 6, w = hw & 63;
    float  v  = qkv[n * QC + DH + c] +
                rowo[((size_t)b * DH + c) * 64 + h] +
                colo[((size_t)b * DH + c) * 64 + w];
    abf[t] = (_Float16)relu6f(v);
}

// ---------------------------------------------------------------------------
extern "C" void kernel_launch(void* const* d_in, const int* in_sizes, int n_in,
                              void* d_out, int out_size, void* d_ws, size_t ws_size,
                              hipStream_t stream)
{
    const float* x      = (const float*)d_in[0];
    const float* Wq     = (const float*)d_in[1];
    const float* bq     = (const float*)d_in[2];
    const float* Wk     = (const float*)d_in[3];
    const float* bk     = (const float*)d_in[4];
    const float* Wv     = (const float*)d_in[5];
    const float* bv     = (const float*)d_in[6];
    const float* W_dw   = (const float*)d_in[7];
    const float* b_dw   = (const float*)d_in[8];
    const float* W_pw   = (const float*)d_in[9];
    const float* b_pw   = (const float*)d_in[10];
    const float* W_row  = (const float*)d_in[11];
    const float* b_row  = (const float*)d_in[12];
    const float* W_col  = (const float*)d_in[13];
    const float* b_col  = (const float*)d_in[14];
    const float* W_proj = (const float*)d_in[15];
    const float* b_proj = (const float*)d_in[16];
    const float* pe_rq  = (const float*)d_in[17];
    const float* pe_rk  = (const float*)d_in[18];
    const float* pe_cq  = (const float*)d_in[19];
    const float* pe_ck  = (const float*)d_in[20];

    char*  ws  = (char*)d_ws;
    size_t off = 0;
    auto alloc = [&](size_t bytes) -> void* {
        void* p = (void*)(ws + off);
        off += (bytes + 255) & ~(size_t)255;
        return p;
    };

    _Float16* xbf     = (_Float16*)alloc((size_t)NPIX * DIM * 2);
    _Float16* wqkv    = (_Float16*)alloc((size_t)DIM * QC * 2);
    float*    bias512 = (float*)   alloc((size_t)QC * 4);
    float*    qkv     = (float*)   alloc((size_t)NPIX * QC * 4);
    _Float16* hbf     = (_Float16*)alloc((size_t)NPIX * QC * 2);
    _Float16* wpw     = (_Float16*)alloc((size_t)QC * DIM * 2);
    float*    pwout   = (float*)   alloc((size_t)NPIX * DIM * 4);
    float*    mrow    = (float*)   alloc((size_t)Bn * QC * 64 * 4);
    float*    mcol    = (float*)   alloc((size_t)Bn * QC * 64 * 4);
    float*    perq    = (float*)   alloc((size_t)128 * 64 * 4);
    float*    perk    = (float*)   alloc((size_t)128 * 64 * 4);
    float*    pecq    = (float*)   alloc((size_t)128 * 64 * 4);
    float*    peck    = (float*)   alloc((size_t)128 * 64 * 4);
    float*    xxrow   = (float*)   alloc((size_t)Bn * DH * 64 * 4);
    float*    xxcol   = (float*)   alloc((size_t)Bn * DH * 64 * 4);
    float*    rowo    = (float*)   alloc((size_t)Bn * DH * 64 * 4);
    float*    colo    = (float*)   alloc((size_t)Bn * DH * 64 * 4);
    _Float16* abf     = (_Float16*)alloc((size_t)NPIX * DH * 2);
    _Float16* wproj   = (_Float16*)alloc((size_t)DH * DIM * 2);

    // 1. conversions / packing
    k_convert_x<<<(NPIX * DIM) / 256, 256, 0, stream>>>(x, xbf);
    k_pack_wqkv<<<(QC * DIM) / 256, 256, 0, stream>>>(Wq, Wk, Wv, wqkv);
    k_pack_bias<<<2, 256, 0, stream>>>(bq, bk, bv, bias512);
    k_pack_w<<<(DIM * QC + 255) / 256, 256, 0, stream>>>(W_pw, wpw, DIM, QC);
    k_pack_w<<<(DIM * DH + 255) / 256, 256, 0, stream>>>(W_proj, wproj, DIM, DH);

    // 2. QKV conv1x1 as WMMA GEMM: qkv[n,512] = x[n,256] x wqkv + bias
    {
        dim3 g(QC / 128, NPIX / 16);
        gemm_wmma_f16<0><<<g, 256, 0, stream>>>(xbf, wqkv, bias512, qkv, nullptr,
                                                NPIX, QC, DIM);
    }

    // 3. depthwise 3x3 + relu6 -> f16
    k_dwconv<<<(int)(((size_t)NPIX * QC) / 256), 256, 0, stream>>>(qkv, W_dw, b_dw, hbf);

    // 4. pointwise conv as WMMA GEMM: pwout[n,256] = hbf[n,512] x wpw + b_pw
    {
        dim3 g(DIM / 128, NPIX / 16);
        gemm_wmma_f16<0><<<g, 256, 0, stream>>>(hbf, wpw, b_pw, pwout, nullptr,
                                                NPIX, DIM, QC);
    }

    // 5. axial attention path (tiny)
    k_mean<<<(Bn * QC * 64) / 256, 256, 0, stream>>>(qkv, mrow, mcol);
    k_peinterp<<<32, 256, 0, stream>>>(pe_rq, perq);
    k_peinterp<<<32, 256, 0, stream>>>(pe_rk, perk);
    k_peinterp<<<32, 256, 0, stream>>>(pe_cq, pecq);
    k_peinterp<<<32, 256, 0, stream>>>(pe_ck, peck);
    k_axial<<<Bn * NH, 64, 0, stream>>>(mrow, perq, perk, xxrow);
    k_axial<<<Bn * NH, 64, 0, stream>>>(mcol, pecq, peck, xxcol);
    k_smallconv<<<(Bn * DH * 64) / 256, 256, 0, stream>>>(xxrow, W_row, b_row, rowo);
    k_smallconv<<<(Bn * DH * 64) / 256, 256, 0, stream>>>(xxcol, W_col, b_col, colo);

    // 6. fuse v + row + col, relu6, downconvert for proj GEMM
    k_aprep<<<(int)(((size_t)NPIX * DH) / 256), 256, 0, stream>>>(qkv, rowo, colo, abf);

    // 7. proj conv + h_sigmoid * pwout, scattered to NCHW output (fused epilogue)
    {
        dim3 g(DIM / 128, NPIX / 16);
        gemm_wmma_f16<1><<<g, 256, 0, stream>>>(abf, wproj, b_proj, (float*)d_out,
                                                pwout, NPIX, DIM, DH);
    }
}
